// PrototypeClassifier_9182640079462
// MI455X (gfx1250) — matrix operand order repaired and checked by reading, not compiled, not run on hardware
//
#include <hip/hip_runtime.h>
#include <hip/hip_bf16.h>
#include <math.h>

// ---------------------------------------------------------------------------
// Problem dimensions (fixed by the reference)
// ---------------------------------------------------------------------------
constexpr int Bn = 8192;   // batch
constexpr int Pn = 2048;   // prototypes
constexpr int Dn = 512;    // feature dim / output dim

typedef __attribute__((ext_vector_type(16))) __bf16 v16bf;
typedef __attribute__((ext_vector_type(8)))  __bf16 v8bf;
typedef __attribute__((ext_vector_type(8)))  float  v8f;

// ---------------------------------------------------------------------------
// Helpers
// ---------------------------------------------------------------------------
__device__ __forceinline__ v8bf ld8(const __bf16* p) {
    return *reinterpret_cast<const v8bf*>(p);
}

__device__ __forceinline__ v16bf cat16(v8bf a, v8bf b) {
    return __builtin_shufflevector(a, b, 0,1,2,3,4,5,6,7,8,9,10,11,12,13,14,15);
}

// A-matrix fragment (16x32 bf16, row-major source [rows, ld], K contiguous).
// ISA layout: lanes 0-15 (M = lane) hold K = {k0..k0+7, k0+16..k0+23};
//             lanes 16-31 hold K = {k0+8..k0+15, k0+24..k0+31}.
__device__ __forceinline__ v16bf load_a_frag(const __bf16* A, int ld,
                                             int row0, int k0, int lane) {
    int r  = row0 + (lane & 15);
    int kk = k0 + ((lane >> 4) << 3);                 // +8 for upper half-wave
    const __bf16* p = A + (size_t)r * (size_t)ld + kk;
    return cat16(ld8(p), ld8(p + 16));
}

// B-matrix fragment (32x16 bf16, but source stored as [N rows, K] row-major
// i.e. the NT-GEMM case: K contiguous per N row).
// ISA layout: lane = N column; lanes 0-15 hold K = {k0..k0+7, k0+16..k0+23};
//             lanes 16-31 hold K = {k0+8..k0+15, k0+24..k0+31}.
__device__ __forceinline__ v16bf load_b_frag(const __bf16* B, int ld,
                                             int col0, int k0, int lane) {
    int c  = col0 + (lane & 15);
    int kk = k0 + ((lane >> 4) << 3);                 // +8 for upper half-wave
    const __bf16* p = B + (size_t)c * (size_t)ld + kk;
    return cat16(ld8(p), ld8(p + 16));
}

__device__ __forceinline__ v8f wmma_bf16(v16bf a, v16bf b, v8f c) {
    return __builtin_amdgcn_wmma_f32_16x16x32_bf16(
        /*neg_a=*/false, a, /*neg_b=*/false, b,
        /*c_mod=*/(short)0, c, /*reuse_a=*/false, /*reuse_b=*/false);
}

// ---------------------------------------------------------------------------
// Kernel 1: split f32 rows into bf16 hi/lo planes + optional row squared-norm
// One 256-thread block per row.
// ---------------------------------------------------------------------------
__global__ __launch_bounds__(256)
void split_rows_kernel(const float* __restrict__ src, int cols,
                       __bf16* __restrict__ hi, __bf16* __restrict__ lo,
                       float* __restrict__ sq) {
    const int row = blockIdx.x;
    const size_t base = (size_t)row * (size_t)cols;
    float partial = 0.0f;
    for (int c = threadIdx.x; c < cols; c += 256) {
        float x  = src[base + c];
        __bf16 h = (__bf16)x;
        __bf16 l = (__bf16)(x - (float)h);
        hi[base + c] = h;
        lo[base + c] = l;
        partial += x * x;
    }
    if (sq != nullptr) {
        __shared__ float red[256];
        red[threadIdx.x] = partial;
        __syncthreads();
        for (int s = 128; s > 0; s >>= 1) {
            if (threadIdx.x < s) red[threadIdx.x] += red[threadIdx.x + s];
            __syncthreads();
        }
        if (threadIdx.x == 0) sq[row] = red[0];
    }
}

// ---------------------------------------------------------------------------
// Kernel 2: bf16x3 NT-GEMM. C = A * B^T with A:[M,K], B:[N,K], K contiguous.
//   EPI == 0 : dist epilogue  d = sqrt(max(rowsq + colsq - 2c, 0)),
//              re-split into bf16 hi/lo planes (feeds GEMM2).
//   EPI == 1 : out epilogue   o = c + bias[n], f32 store.
// Block = 128 threads = 4 waves stacked along M; wave tile = 64x64.
// Per K-step(32): 48 v_wmma_f32_16x16x32_bf16 vs 32 global_load_b128.
// ---------------------------------------------------------------------------
template <int EPI>
__global__ __launch_bounds__(128)
void gemm_bf16x3_kernel(const __bf16* __restrict__ Ahi,
                        const __bf16* __restrict__ Alo,
                        const __bf16* __restrict__ Bhi,
                        const __bf16* __restrict__ Blo,
                        int M, int N, int K,
                        const float* __restrict__ rowsq,
                        const float* __restrict__ colsq,
                        __bf16* __restrict__ Dhi,
                        __bf16* __restrict__ Dlo,
                        const float* __restrict__ bias,
                        float* __restrict__ Out) {
    const int lane = threadIdx.x & 31;
    const int wave = threadIdx.x >> 5;
    const int m0 = (blockIdx.x * 4 + wave) * 64;   // 4 waves along M
    const int n0 = blockIdx.y * 64;

    v8f acc[4][4];
#pragma unroll
    for (int mi = 0; mi < 4; ++mi)
#pragma unroll
        for (int ni = 0; ni < 4; ++ni)
            acc[mi][ni] = (v8f)(0.0f);

    for (int k0 = 0; k0 < K; k0 += 32) {
        v16bf ah[4], al[4];
#pragma unroll
        for (int mi = 0; mi < 4; ++mi) {
            ah[mi] = load_a_frag(Ahi, K, m0 + mi * 16, k0, lane);
            al[mi] = load_a_frag(Alo, K, m0 + mi * 16, k0, lane);
        }
#pragma unroll
        for (int ni = 0; ni < 4; ++ni) {
            v16bf bh = load_b_frag(Bhi, K, n0 + ni * 16, k0, lane);
            v16bf bl = load_b_frag(Blo, K, n0 + ni * 16, k0, lane);
#pragma unroll
            for (int mi = 0; mi < 4; ++mi) {
                // bf16x3: hi*hi + lo*hi + hi*lo  (lo*lo dropped, < 2^-16 rel)
                acc[mi][ni] = wmma_bf16(ah[mi], bh, acc[mi][ni]);
                acc[mi][ni] = wmma_bf16(al[mi], bh, acc[mi][ni]);
                acc[mi][ni] = wmma_bf16(ah[mi], bl, acc[mi][ni]);
            }
        }
    }

    // Epilogue. C/D layout: vgpr i of lanes 0-15 is (M = i,   N = lane),
    //                       vgpr i of lanes 16-31 is (M = i+8, N = lane-16).
    const int moff = (lane < 16) ? 0 : 8;
    const int nc   = lane & 15;
#pragma unroll
    for (int mi = 0; mi < 4; ++mi) {
#pragma unroll
        for (int ni = 0; ni < 4; ++ni) {
#pragma unroll
            for (int i = 0; i < 8; ++i) {
                const int m = m0 + mi * 16 + i + moff;
                const int n = n0 + ni * 16 + nc;
                const float c = acc[mi][ni][i];
                if (EPI == 0) {
                    float d2 = rowsq[m] + colsq[n] - 2.0f * c;
                    float d  = sqrtf(fmaxf(d2, 0.0f));
                    __bf16 h = (__bf16)d;
                    __bf16 l = (__bf16)(d - (float)h);
                    const size_t idx = (size_t)m * (size_t)N + n;
                    Dhi[idx] = h;
                    Dlo[idx] = l;
                } else {
                    Out[(size_t)m * (size_t)N + n] = c + bias[n];
                }
            }
        }
    }
}

// ---------------------------------------------------------------------------
// Launcher
// ---------------------------------------------------------------------------
extern "C" void kernel_launch(void* const* d_in, const int* in_sizes, int n_in,
                              void* d_out, int out_size, void* d_ws, size_t ws_size,
                              hipStream_t stream) {
    (void)in_sizes; (void)n_in; (void)out_size; (void)ws_size;

    const float* x     = (const float*)d_in[0];   // [Bn, Dn]
    const float* proto = (const float*)d_in[1];   // [Pn, Dn]
    const float* W     = (const float*)d_in[2];   // [Dn, Pn]
    const float* bias  = (const float*)d_in[3];   // [Dn]
    float* out         = (float*)d_out;           // [Bn, Dn]

    // Workspace carve-up (256B aligned blocks).
    size_t off = 0;
    auto carve = [&](size_t bytes) -> void* {
        void* p = (void*)((char*)d_ws + off);
        off += (bytes + 255) & ~(size_t)255;
        return p;
    };
    __bf16* Xhi = (__bf16*)carve((size_t)Bn * Dn * 2);
    __bf16* Xlo = (__bf16*)carve((size_t)Bn * Dn * 2);
    __bf16* Phi = (__bf16*)carve((size_t)Pn * Dn * 2);
    __bf16* Plo = (__bf16*)carve((size_t)Pn * Dn * 2);
    __bf16* Whi = (__bf16*)carve((size_t)Dn * Pn * 2);
    __bf16* Wlo = (__bf16*)carve((size_t)Dn * Pn * 2);
    __bf16* Ghi = (__bf16*)carve((size_t)Bn * Pn * 2);   // dist hi
    __bf16* Glo = (__bf16*)carve((size_t)Bn * Pn * 2);   // dist lo
    float*  insq = (float*)carve((size_t)Bn * 4);
    float*  prsq = (float*)carve((size_t)Pn * 4);

    // 1) f32 -> bf16 hi/lo splits + row squared norms.
    split_rows_kernel<<<Bn, 256, 0, stream>>>(x,     Dn, Xhi, Xlo, insq);
    split_rows_kernel<<<Pn, 256, 0, stream>>>(proto, Dn, Phi, Plo, prsq);
    split_rows_kernel<<<Dn, 256, 0, stream>>>(W,     Pn, Whi, Wlo, nullptr);

    // 2) dist = sqrt(max(|x|^2 + |p|^2 - 2 X.P^T, 0))   [Bn, Pn]
    //    M=8192, N=2048, K=512 -> grid (8192/256, 2048/64)
    gemm_bf16x3_kernel<0><<<dim3(Bn / 256, Pn / 64), 128, 0, stream>>>(
        Xhi, Xlo, Phi, Plo, Bn, Pn, Dn,
        insq, prsq, Ghi, Glo, nullptr, nullptr);

    // 3) out = dist @ W^T + b                            [Bn, Dn]
    //    M=8192, N=512, K=2048 -> grid (8192/256, 512/64)
    gemm_bf16x3_kernel<1><<<dim3(Bn / 256, Dn / 64), 128, 0, stream>>>(
        Ghi, Glo, Whi, Wlo, Bn, Dn, Pn,
        nullptr, nullptr, nullptr, nullptr, bias, out);
}
